// SpikingPolicyNetwork_90812788506855
// MI455X (gfx1250) — matrix operand order repaired and checked by reading, not compiled, not run on hardware
//
#include <hip/hip_runtime.h>
#include <hip/hip_bf16.h>

typedef _Float16 v8h  __attribute__((ext_vector_type(8)));
typedef _Float16 v16h __attribute__((ext_vector_type(16)));
typedef float    v8f  __attribute__((ext_vector_type(8)));

constexpr int  BATCH = 4096;
constexpr int  INDIM = 512;
constexpr int  HID   = 1024;
constexpr int  OUTD  = 256;
constexpr int  TSTEP = 16;
constexpr float DECAY = 1.0f - 1.0f / 20.0f;   // 0.95
constexpr float V_TH  = 1.0f;

// ---------------------------------------------------------------------------
// f32 -> f16 convert
// ---------------------------------------------------------------------------
__global__ __launch_bounds__(256) void cvt_f16_kernel(const float* __restrict__ src,
                                                      _Float16* __restrict__ dst, int n) {
    int i = blockIdx.x * 256 + threadIdx.x;
    if (i < n) dst[i] = (_Float16)src[i];
}

// ---------------------------------------------------------------------------
// Fused WMMA GEMM + LIF.
//   A: [M x K] row-major f16 (x or spikes)      W: [N x K] row-major f16
//   D[m,n] = sum_k A[m,k]*W[n,k] + bias[n]
// MODE 0: store current I (=acc+bias) into Vws (fragment-major)  [layer-0 prep]
// MODE 1: LIF update on Vws, write spikes f16 row-major to Sout  [layer 1]
// MODE 2: LIF update on Vws, accumulate spikes into Out f32      [layer 2]
// Block: 256 thr = 8 waves (wave32), arranged 2(M) x 4(N).
// Wave tile: (MT*16) x 64  -> MT*4 WMMA accumulators; B fragments reused
// across the MT A fragments to cut per-WMMA load bytes.
// Block tile: (MT*32) x 256.
// ---------------------------------------------------------------------------
template <int MODE, int MT>
__global__ __launch_bounds__(256) void gemm_lif_kernel(
    const _Float16* __restrict__ A, const _Float16* __restrict__ W,
    const float* __restrict__ bias, float* __restrict__ Vws,
    _Float16* __restrict__ Sout, float* __restrict__ Out,
    int M, int N, int K, int first) {

    const int lane  = threadIdx.x & 31;
    const int wave  = threadIdx.x >> 5;
    const int l16   = lane & 15;
    const int lhalf = lane >> 4;           // 0: lanes 0-15, 1: lanes 16-31

    const int m0 = blockIdx.x * (MT * 32) + (wave >> 2) * (MT * 16);  // wave's M origin
    const int nW = blockIdx.y * 256 + (wave & 3) * 64;                // wave's N origin

    // A fragment (16x32 f16): lane holds row m0+mt*16+l16, K = k + lhalf*8 (+0..7) and +16
    const _Float16* aRow[MT];
#pragma unroll
    for (int mt = 0; mt < MT; ++mt)
        aRow[mt] = A + (size_t)(m0 + mt * 16 + l16) * K + lhalf * 8;
    // B fragment (32x16 f16): lane holds col nW+16j+l16, K = k + lhalf*8 (+0..7) and +16
    const _Float16* bRow = W + (size_t)(nW + l16) * K + lhalf * 8;

    v8f acc[MT][4];
#pragma unroll
    for (int mt = 0; mt < MT; ++mt)
#pragma unroll
        for (int j = 0; j < 4; ++j) acc[mt][j] = (v8f){};

    for (int k = 0; k < K; k += 32) {
        // keep the W stream flowing ahead (weights are L2-resident)
        __builtin_prefetch(bRow + k + 512, 0, 1);

        v16h a[MT];
#pragma unroll
        for (int mt = 0; mt < MT; ++mt) {
            v8h alo = *(const v8h*)(aRow[mt] + k);
            v8h ahi = *(const v8h*)(aRow[mt] + k + 16);
            a[mt] = __builtin_shufflevector(alo, ahi,
                     0,1,2,3,4,5,6,7,8,9,10,11,12,13,14,15);
        }

#pragma unroll
        for (int j = 0; j < 4; ++j) {
            const _Float16* bp = bRow + (size_t)(16 * j) * K + k;
            v8h blo = *(const v8h*)(bp);
            v8h bhi = *(const v8h*)(bp + 16);
            v16h b = __builtin_shufflevector(blo, bhi,
                      0,1,2,3,4,5,6,7,8,9,10,11,12,13,14,15);
#pragma unroll
            for (int mt = 0; mt < MT; ++mt)
                acc[mt][j] = __builtin_amdgcn_wmma_f32_16x16x32_f16(
                    false, a[mt], false, b, (short)0, acc[mt][j], false, false);
        }
    }

    const int tilesPerRow = N >> 4;
#pragma unroll
    for (int mt = 0; mt < MT; ++mt) {
        const int mTile = m0 + mt * 16;
#pragma unroll
        for (int j = 0; j < 4; ++j) {
            const int nTile = nW + 16 * j;
            const int n     = nTile + l16;
            const float bn  = bias[n];
            const size_t tileId = (size_t)(mTile >> 4) * tilesPerRow + (nTile >> 4);
            float* vp = Vws + tileId * 256 + (size_t)lane * 8;   // contiguous 32B/lane

            if constexpr (MODE == 0) {
#pragma unroll
                for (int r = 0; r < 8; ++r) vp[r] = acc[mt][j][r] + bn;
            } else {
#pragma unroll
                for (int r = 0; r < 8; ++r) {
                    float cur  = acc[mt][j][r] + bn;
                    float vold = first ? 0.0f : vp[r];
                    float v    = DECAY * vold + cur;
                    bool  fire = (v >= V_TH);
                    vp[r] = fire ? 0.0f : v;                      // reset to 0
                    float sp = fire ? 1.0f : 0.0f;
                    const size_t row = (size_t)(mTile + lhalf * 8 + r);
                    if constexpr (MODE == 1) {
                        Sout[row * N + n] = (_Float16)sp;
                    } else {
                        float* op = Out + row * N + n;
                        if (first) *op = sp; else *op += sp;
                    }
                }
            }
        }
    }
}

// ---------------------------------------------------------------------------
// Layer-0 elementwise LIF: v0 = DECAY*v0 + I0 ; spike ; reset.
// I0 / V0 are fragment-major; spikes written row-major f16 for layer-1 WMMA A.
// One thread handles one lane-fragment of 8 consecutive floats.
// ---------------------------------------------------------------------------
__global__ __launch_bounds__(256) void lif0_kernel(const float* __restrict__ I0,
                                                   float* __restrict__ V0,
                                                   _Float16* __restrict__ S0, int first) {
    const size_t frag = (size_t)blockIdx.x * 256 + threadIdx.x;  // 8-elem fragment id
    const size_t base = frag * 8;
    const size_t tileId = base >> 8;
    const int lane = (int)((base >> 3) & 31);
    const int tilesPerRow = HID >> 4;                 // 64
    const int m0 = (int)(tileId / tilesPerRow) * 16;
    const int n0 = (int)(tileId % tilesPerRow) * 16;
    const int mBase = m0 + (lane >> 4) * 8;
    const int n = n0 + (lane & 15);

#pragma unroll
    for (int r = 0; r < 8; ++r) {
        float cur  = I0[base + r];
        float vold = first ? 0.0f : V0[base + r];
        float v    = DECAY * vold + cur;
        bool fire  = (v >= V_TH);
        V0[base + r] = fire ? 0.0f : v;
        S0[(size_t)(mBase + r) * HID + n] = (_Float16)(fire ? 1.0f : 0.0f);
    }
}

// ---------------------------------------------------------------------------
// Host launcher
// ---------------------------------------------------------------------------
extern "C" void kernel_launch(void* const* d_in, const int* in_sizes, int n_in,
                              void* d_out, int out_size, void* d_ws, size_t ws_size,
                              hipStream_t stream) {
    const float* x  = (const float*)d_in[0];
    const float* W0 = (const float*)d_in[1];
    const float* b0 = (const float*)d_in[2];
    const float* W1 = (const float*)d_in[3];
    const float* b1 = (const float*)d_in[4];
    const float* W2 = (const float*)d_in[5];
    const float* b2 = (const float*)d_in[6];
    float* out = (float*)d_out;

    // workspace carve-up (all offsets 256B aligned)
    char* ws = (char*)d_ws;
    size_t off = 0;
    auto carve = [&](size_t bytes) { void* p = ws + off; off += (bytes + 255) & ~(size_t)255; return p; };
    _Float16* xh  = (_Float16*)carve((size_t)BATCH * INDIM * 2);   // 4 MiB
    _Float16* w0h = (_Float16*)carve((size_t)HID * INDIM * 2);     // 1 MiB
    _Float16* w1h = (_Float16*)carve((size_t)HID * HID * 2);       // 2 MiB
    _Float16* w2h = (_Float16*)carve((size_t)OUTD * HID * 2);      // 0.5 MiB
    float* I0 = (float*)carve((size_t)BATCH * HID * 4);            // 16 MiB
    float* V0 = (float*)carve((size_t)BATCH * HID * 4);            // 16 MiB
    float* V1 = (float*)carve((size_t)BATCH * HID * 4);            // 16 MiB
    float* V2 = (float*)carve((size_t)BATCH * OUTD * 4);           // 4 MiB
    _Float16* S0 = (_Float16*)carve((size_t)BATCH * HID * 2);      // 8 MiB
    _Float16* S1 = (_Float16*)carve((size_t)BATCH * HID * 2);      // 8 MiB
    (void)ws_size; (void)in_sizes; (void)n_in; (void)out_size;

    // 1) precision converts
    {
        int n;
        n = BATCH * INDIM; cvt_f16_kernel<<<(n + 255) / 256, 256, 0, stream>>>(x,  xh,  n);
        n = HID * INDIM;   cvt_f16_kernel<<<(n + 255) / 256, 256, 0, stream>>>(W0, w0h, n);
        n = HID * HID;     cvt_f16_kernel<<<(n + 255) / 256, 256, 0, stream>>>(W1, w1h, n);
        n = OUTD * HID;    cvt_f16_kernel<<<(n + 255) / 256, 256, 0, stream>>>(W2, w2h, n);
    }

    // 2) time-invariant input current: I0 = x @ W0^T + b0 (computed once)
    gemm_lif_kernel<0, 2><<<dim3(BATCH / 64, HID / 256), 256, 0, stream>>>(
        xh, w0h, b0, I0, nullptr, nullptr, BATCH, HID, INDIM, 0);

    // 3) unrolled timestep loop
    for (int t = 0; t < TSTEP; ++t) {
        const int first = (t == 0) ? 1 : 0;
        // layer 0 LIF (elementwise)
        lif0_kernel<<<(BATCH * HID / 8) / 256, 256, 0, stream>>>(I0, V0, S0, first);
        // layer 1: s0 @ W1^T + b1 -> LIF -> s1   (dominant GEMM, fat 32x64 wave tile)
        gemm_lif_kernel<1, 2><<<dim3(BATCH / 64, HID / 256), 256, 0, stream>>>(
            S0, w1h, b1, V1, S1, nullptr, BATCH, HID, HID, first);
        // layer 2: s1 @ W2^T + b2 -> LIF -> accumulate spikes into out
        // (MT=1 keeps 128 blocks in flight for the narrow N=256 output)
        gemm_lif_kernel<2, 1><<<dim3(BATCH / 32, OUTD / 256), 256, 0, stream>>>(
            S1, w2h, b2, V2, nullptr, out, BATCH, OUTD, HID, first);
    }
}
